// AttentionBlock_14319420965129
// MI455X (gfx1250) — matrix operand order repaired and checked by reading, not compiled, not run on hardware
//
#include <hip/hip_runtime.h>
#include <hip/hip_bf16.h>
#include <math.h>

// Problem constants (from reference)
#define IN_CH 512
#define KSZ   64
#define NB    4
#define TT    4096
#define OUTC  576   // 512 + 64
#define PCOL  192   // concatenated projection columns: q|k|v

typedef float    v2f  __attribute__((ext_vector_type(2)));
typedef float    v8f  __attribute__((ext_vector_type(8)));
typedef _Float16 v8h  __attribute__((ext_vector_type(8)));
typedef _Float16 v16h __attribute__((ext_vector_type(16)));

static __device__ inline v16h cat16(v8h lo, v8h hi) {
  return __builtin_shufflevector(lo, hi, 0,1,2,3,4,5,6,7,8,9,10,11,12,13,14,15);
}

// ---------------------------------------------------------------------------
// Kernel 0: concatenate Wq|Wk|Wv -> Wcat[512][192], bq|bk|bv -> bcat[192].
// The 1/sqrt(64) logit scale is folded into the q columns and bq.
// ---------------------------------------------------------------------------
__global__ __launch_bounds__(256)
void concat_w_kernel(const float* __restrict__ Wq, const float* __restrict__ bq,
                     const float* __restrict__ Wk, const float* __restrict__ bk,
                     const float* __restrict__ Wv, const float* __restrict__ bv,
                     float* __restrict__ Wcat, float* __restrict__ bcat)
{
  const int i = blockIdx.x * 256 + threadIdx.x;
  if (i < IN_CH * PCOL) {
    const int k = i / PCOL, c = i % PCOL;
    float v;
    if (c < 64)       v = Wq[k * KSZ + c] * 0.125f;
    else if (c < 128) v = Wk[k * KSZ + (c - 64)];
    else              v = Wv[k * KSZ + (c - 128)];
    Wcat[i] = v;
  }
  if (i < PCOL) {
    float v;
    if (i < 64)       v = bq[i] * 0.125f;
    else if (i < 128) v = bk[i - 64];
    else              v = bv[i - 128];
    bcat[i] = v;
  }
}

// ---------------------------------------------------------------------------
// Kernel 1: QKV projection (fp32 WMMA 16x16x4) + x -> out[...,0:512] copy.
// B is loaded from the single concatenated Wcat pointer (pure global path).
// Stores q (pre-scaled) and k row-major f16 [N][T][64],
// and v transposed f16 [N][64][T] so kernel 2's B-fragments are contiguous.
// ---------------------------------------------------------------------------
__global__ __launch_bounds__(256)
void qkv_proj_kernel(const float* __restrict__ x,
                     const float* __restrict__ Wcat,
                     const float* __restrict__ bcat,
                     float* __restrict__ out,
                     _Float16* __restrict__ qh, _Float16* __restrict__ kh,
                     _Float16* __restrict__ vt)
{
  const int n    = blockIdx.y;
  const int t0   = blockIdx.x * 64;
  const int tid  = threadIdx.x;
  const int w    = tid >> 5;       // wave id 0..7
  const int lane = tid & 31;
  const int lm   = lane & 15;
  const int lh   = lane >> 4;      // lane-half

  // ---- copy x rows into out channels [0,512)
  {
    const float4* xin = (const float4*)(x + ((size_t)(n*TT + t0)) * IN_CH);
    for (int i = tid; i < 64 * 128; i += 256) {
      int rr = i >> 7, cc = i & 127;
      float4 vv = xin[rr * 128 + cc];
      *(float4*)(out + ((size_t)(n*TT + t0 + rr)) * OUTC + cc * 4) = vv;
    }
  }

  // ---- projections: wave -> (row tile r, column group g); 6 16x16 tiles each
  const int r = w & 3;
  const int g = w >> 2;
  const int rowg = t0 + r * 16 + lm;

  v8f acc[6];
  #pragma unroll
  for (int i = 0; i < 6; i++) acc[i] = (v8f)0.0f;

  const float* xrow = x + ((size_t)(n*TT + rowg)) * IN_CH;
  const int colbase = g * 96 + lm;   // first of this wave's 6 column tiles

  for (int kk = 0; kk < IN_CH; kk += 4) {
    const int k2 = kk + lh * 2;                 // A: lanes<16 hold K=0,1; lanes>=16 K=2,3
    v2f a = *(const v2f*)(xrow + k2);           // 8B-aligned global_load_b64
    const float* w0 = Wcat + (size_t)k2 * PCOL + colbase;
    const float* w1 = Wcat + (size_t)(k2 + 1) * PCOL + colbase;
    #pragma unroll
    for (int i = 0; i < 6; i++) {
      v2f b;                                    // B: lanes<16 K=0,1 ; lanes>=16 K=2,3
      b.x = w0[i * 16];
      b.y = w1[i * 16];
      acc[i] = __builtin_amdgcn_wmma_f32_16x16x4_f32(
          false, a, false, b, (short)0, acc[i], false, false);
    }
  }

  // ---- bias add and store staging buffers (f16); fixed pointer per branch
  #pragma unroll
  for (int i = 0; i < 6; i++) {
    const int ct   = g * 6 + i;        // 0..11 ; /4 -> matrix
    const int col  = ct * 16 + lm;     // column in concatenated space
    const float bias = bcat[col];
    #pragma unroll
    for (int j = 0; j < 8; j++) {
      const int rowt = t0 + r * 16 + j + 8 * lh;  // C-layout: row = vgpr + 8*half
      const float vvv = acc[i][j] + bias;
      if (ct < 4) {
        qh[((size_t)(n*TT + rowt)) * KSZ + col] = (_Float16)vvv;           // already scaled
      } else if (ct < 8) {
        kh[((size_t)(n*TT + rowt)) * KSZ + (col - 64)] = (_Float16)vvv;
      } else {
        vt[((size_t)(n*KSZ + (col - 128))) * TT + rowt] = (_Float16)vvv;   // transposed
      }
    }
  }
}

// ---------------------------------------------------------------------------
// Kernel 2: causal flash attention with online softmax.
// 4 waves / block; each wave owns a 16-row q tile, loops over 32-col chunks.
// S and O via v_wmma_f32_16x16x32_f16; P re-layout C->A through LDS.
// ---------------------------------------------------------------------------
__global__ __launch_bounds__(128)
void flash_attn_kernel(const _Float16* __restrict__ qh,
                       const _Float16* __restrict__ kh,
                       const _Float16* __restrict__ vt,
                       float* __restrict__ out)
{
  __shared__ __align__(32) _Float16 pshare[4][16][32];

  const int n    = blockIdx.y;
  const int t0   = blockIdx.x * 64;
  const int w    = threadIdx.x >> 5;
  const int lane = threadIdx.x & 31;
  const int lm   = lane & 15;
  const int lh   = lane >> 4;
  const int t0w  = t0 + w * 16;
  const size_t nT = (size_t)n * TT;

  // Q A-fragments (16x32 f16 layout: [0..7]=K(lh*8+0..7), [8..15]=K(16+lh*8+0..7))
  const _Float16* qrow = qh + (nT + t0w + lm) * KSZ;
  const v16h a0 = cat16(*(const v8h*)(qrow +      lh * 8), *(const v8h*)(qrow + 16 + lh * 8));
  const v16h a1 = cat16(*(const v8h*)(qrow + 32 + lh * 8), *(const v8h*)(qrow + 48 + lh * 8));

  v8f o[4];
  #pragma unroll
  for (int nt = 0; nt < 4; nt++) o[nt] = (v8f)0.0f;
  float mrow[8], lrow[8];
  #pragma unroll
  for (int j = 0; j < 8; j++) { mrow[j] = -INFINITY; lrow[j] = 0.0f; }

  const float L2E = 1.4426950408889634f;
  const int smax = t0w + 15;

  for (int s0 = 0; s0 <= smax; s0 += 32) {
    // ---- S = (scaled Q) @ K^T for two 16x16 subtiles
    v8f sc[2];
    #pragma unroll
    for (int st = 0; st < 2; st++) {
      const _Float16* krow = kh + (nT + s0 + st * 16 + lm) * KSZ;
      const v16h b0 = cat16(*(const v8h*)(krow +      lh * 8), *(const v8h*)(krow + 16 + lh * 8));  // B: [0..7]=K(lh*8+0..7), [8..15]=K(16+lh*8+0..7)
      const v16h b1 = cat16(*(const v8h*)(krow + 32 + lh * 8), *(const v8h*)(krow + 48 + lh * 8));
      v8f s = (v8f)0.0f;
      s = __builtin_amdgcn_wmma_f32_16x16x32_f16(false, a0, false, b0, (short)0, s, false, false);
      s = __builtin_amdgcn_wmma_f32_16x16x32_f16(false, a1, false, b1, (short)0, s, false, false);
      sc[st] = s;
    }

    // ---- causal mask (C-layout: row = j + 8*lh, col = s0 + st*16 + lm)
    #pragma unroll
    for (int st = 0; st < 2; st++) {
      const int col = s0 + st * 16 + lm;
      #pragma unroll
      for (int j = 0; j < 8; j++) {
        if (col > t0w + j + 8 * lh) sc[st][j] = -INFINITY;
      }
    }

    // ---- online softmax (row reductions across the 16-lane group)
    #pragma unroll
    for (int j = 0; j < 8; j++) {
      const float v0 = sc[0][j], v1 = sc[1][j];
      float cm = fmaxf(v0, v1);
      cm = fmaxf(cm, __shfl_xor(cm, 1, 32));
      cm = fmaxf(cm, __shfl_xor(cm, 2, 32));
      cm = fmaxf(cm, __shfl_xor(cm, 4, 32));
      cm = fmaxf(cm, __shfl_xor(cm, 8, 32));
      const float mn = fmaxf(mrow[j], cm);
      const float al = __builtin_amdgcn_exp2f((mrow[j] - mn) * L2E);
      mrow[j] = mn;
      const float p0 = __builtin_amdgcn_exp2f((v0 - mn) * L2E);
      const float p1 = __builtin_amdgcn_exp2f((v1 - mn) * L2E);
      float rs = p0 + p1;
      rs += __shfl_xor(rs, 1, 32);
      rs += __shfl_xor(rs, 2, 32);
      rs += __shfl_xor(rs, 4, 32);
      rs += __shfl_xor(rs, 8, 32);
      lrow[j] = lrow[j] * al + rs;
      o[0][j] *= al; o[1][j] *= al; o[2][j] *= al; o[3][j] *= al;
      pshare[w][j + 8 * lh][lm]      = (_Float16)p0;
      pshare[w][j + 8 * lh][16 + lm] = (_Float16)p1;
    }

    // wave-local LDS RAW: ensure stores land before cross-lane re-read
    __builtin_amdgcn_wave_barrier();
    asm volatile("s_wait_dscnt 0" ::: "memory");
    __builtin_amdgcn_wave_barrier();

    // ---- P as A-fragment, O += P @ V
    const v16h ap = cat16(*(const v8h*)&pshare[w][lm][lh * 8],
                          *(const v8h*)&pshare[w][lm][16 + lh * 8]);
    #pragma unroll
    for (int nt = 0; nt < 4; nt++) {
      const _Float16* vc = vt + ((size_t)(n*KSZ) + nt * 16 + lm) * TT + s0;
      const v16h bv = cat16(*(const v8h*)(vc + lh * 8), *(const v8h*)(vc + 16 + lh * 8));  // B: K(=s) = [0..7] lh*8+0..7, [8..15] 16+lh*8+0..7
      o[nt] = __builtin_amdgcn_wmma_f32_16x16x32_f16(
          false, ap, false, bv, (short)0, o[nt], false, false);
    }

    // prefetch next K chunk
    if (s0 + 32 <= smax) {
      __builtin_prefetch(kh + (nT + s0 + 32 + lane) * KSZ, 0, 1);
    }
  }

  // ---- normalize and write out[..., 512:576]
  #pragma unroll
  for (int nt = 0; nt < 4; nt++) {
    #pragma unroll
    for (int j = 0; j < 8; j++) {
      const int row = t0w + j + 8 * lh;
      out[(nT + row) * OUTC + 512 + nt * 16 + lm] = o[nt][j] / lrow[j];
    }
  }
}

// ---------------------------------------------------------------------------
extern "C" void kernel_launch(void* const* d_in, const int* in_sizes, int n_in,
                              void* d_out, int out_size, void* d_ws, size_t ws_size,
                              hipStream_t stream) {
  const float* x  = (const float*)d_in[0];
  const float* Wq = (const float*)d_in[1];
  const float* bq = (const float*)d_in[2];
  const float* Wk = (const float*)d_in[3];
  const float* bk = (const float*)d_in[4];
  const float* Wv = (const float*)d_in[5];
  const float* bv = (const float*)d_in[6];
  float* out = (float*)d_out;

  const size_t qkv_elems = (size_t)NB * TT * KSZ;   // 1M elems = 2MB f16 each
  _Float16* qh = (_Float16*)d_ws;
  _Float16* kh = qh + qkv_elems;
  _Float16* vt = kh + qkv_elems;
  float* Wcat = (float*)(vt + qkv_elems);           // 512*192 floats = 384KB
  float* bcat = Wcat + (size_t)IN_CH * PCOL;        // 192 floats

  concat_w_kernel<<<(IN_CH * PCOL + 255) / 256, 256, 0, stream>>>(
      Wq, bq, Wk, bk, Wv, bv, Wcat, bcat);

  dim3 grid(TT / 64, NB);
  qkv_proj_kernel<<<grid, 256, 0, stream>>>(x, Wcat, bcat, out, qh, kh, vt);
  flash_attn_kernel<<<grid, 128, 0, stream>>>(qh, kh, vt, out);
}
